// TextClassifier_69776038690817
// MI455X (gfx1250) — matrix operand order repaired and checked by reading, not compiled, not run on hardware
//
#include <hip/hip_runtime.h>
#include <hip/hip_bf16.h>
#include <stdint.h>

// ---------------------------------------------------------------------------
// Types for CDNA5 WMMA
// ---------------------------------------------------------------------------
typedef __bf16  v16bf __attribute__((ext_vector_type(16)));
typedef __bf16  v8bf  __attribute__((ext_vector_type(8)));
typedef float   v8f   __attribute__((ext_vector_type(8)));

#define H_   512
#define W_   4096
#define L_   16
#define E_   1024
#define G_   2048   // 4*H
#define T_   4096   // outer sequence length == W
#define K_   512    // contraction dim of both WMMA GEMMs

__device__ __forceinline__ float sigmoidf_(float x) {
    return 1.0f / (1.0f + __expf(-x));
}

__device__ __forceinline__ float dot8(v8bf a, v8bf b) {
    float s = 0.f;
#pragma unroll
    for (int i = 0; i < 8; ++i) s += (float)a[i] * (float)b[i];
    return s;
}

// A 16x32 bf16 fragment: lanes 0-15 hold K {0..15},
// lanes 16-31 hold K {16..31}; per-lane = two contiguous 16B chunks.
__device__ __forceinline__ v16bf load_afrag(const __bf16* rowp, int kh) {
    union { v8bf h[2]; v16bf v; } u;
    u.h[0] = *(const v8bf*)(rowp + kh * 16);
    u.h[1] = *(const v8bf*)(rowp + kh * 16 + 8);
    return u.v;
}

// ---------------------------------------------------------------------------
// fp32 -> bf16 conversion (weights prep; done once per launch)
// ---------------------------------------------------------------------------
__global__ void cvt_f32_bf16(const float* __restrict__ src, __bf16* __restrict__ dst, int n) {
    for (int i = blockIdx.x * blockDim.x + threadIdx.x; i < n; i += gridDim.x * blockDim.x)
        dst[i] = (__bf16)src[i];
}

// ---------------------------------------------------------------------------
// Fused inner word-LSTM step:
//   gates = Hin @ Wh^T  (WMMA bf16, fp32 acc)  then per-element LSTM update.
// Wave computes the SAME 16-column j-range across all 4 gate blocks
// (acc[g] <-> gate g, columns g*512 + [j0..j0+15]) so i/f/g/o live in the
// wave's own registers: no gates round-trip through memory.
// 2-stage software pipeline: K-slab k+1 fragments load while slab k MMAs issue.
// Block: 256 thr = 8 waves as 2(M) x 4(j); tile 32 rows x 64 j-cols.
// Grid: (512/64, 4096/32) = (8, 128).
// ---------------------------------------------------------------------------
__global__ __launch_bounds__(256) void wmma_lstm_step(
    const __bf16* __restrict__ Hin,   // [W,512] bf16 (prev hidden, ping)
    const __bf16* __restrict__ Bw,    // Wh bf16 [2048,512] row-major
    const int* __restrict__ chars, const int* __restrict__ wlens,
    const float* __restrict__ wWi, const float* __restrict__ wb,
    float* __restrict__ Cst,          // [W,512] fp32 cell state (in place)
    __bf16* __restrict__ Hout,        // [W,512] bf16 (pong)
    __bf16* __restrict__ lastbf,      // [W,512] bf16 captured at t==len-1
    int t)
{
    const int lane = threadIdx.x & 31;
    const int wave = threadIdx.x >> 5;
    const int wm   = wave >> 2;          // 0..1
    const int wn   = wave & 3;           // 0..3
    const int m0   = blockIdx.y * 32 + wm * 16;
    const int j0   = blockIdx.x * 64 + wn * 16;
    const int row  = lane & 15;
    const int kh   = lane >> 4;

    const __bf16* Arow = Hin + (size_t)(m0 + row) * K_;
    const __bf16* Brow[4];
#pragma unroll
    for (int g = 0; g < 4; ++g)
        Brow[g] = Bw + (size_t)(g * H_ + j0 + row) * K_ + kh * 16;

    v8f acc[4] = {};

    // ---- 2-stage pipelined K loop ----
    v16bf a_cur = load_afrag(Arow, kh);
    v16bf b_cur[4];
#pragma unroll
    for (int g = 0; g < 4; ++g) b_cur[g] = *(const v16bf*)(Brow[g]);

#pragma unroll
    for (int kb = 0; kb < K_; kb += 32) {
        v16bf a_nxt = a_cur;
        v16bf b_nxt[4];
        if (kb + 32 < K_) {
            a_nxt = load_afrag(Arow + kb + 32, kh);
#pragma unroll
            for (int g = 0; g < 4; ++g) b_nxt[g] = *(const v16bf*)(Brow[g] + kb + 32);
        } else {
#pragma unroll
            for (int g = 0; g < 4; ++g) b_nxt[g] = b_cur[g];
        }
#pragma unroll
        for (int g = 0; g < 4; ++g)
            acc[g] = __builtin_amdgcn_wmma_f32_16x16x32_bf16(
                false, a_cur, false, b_cur[g], (short)0, acc[g], false, false);
        a_cur = a_nxt;
#pragma unroll
        for (int g = 0; g < 4; ++g) b_cur[g] = b_nxt[g];
    }

    // -------- fused LSTM epilogue (all gates in-register) --------
    const int col = j0 + row;            // hidden unit index, fixed per lane
    const float wi_i = wWi[col],          bb_i = wb[col];
    const float wi_f = wWi[H_ + col],     bb_f = wb[H_ + col];
    const float wi_g = wWi[2 * H_ + col], bb_g = wb[2 * H_ + col];
    const float wi_o = wWi[3 * H_ + col], bb_o = wb[3 * H_ + col];

#pragma unroll
    for (int r = 0; r < 8; ++r) {
        const int m = m0 + r + 8 * kh;   // C/D layout: VGPR r -> M = r + 8*(lane>>4)
        const float x = (float)chars[m * L_ + t];
        const float gi = acc[0][r] + wi_i * x + bb_i;
        const float gf = acc[1][r] + wi_f * x + bb_f;
        const float gg = acc[2][r] + wi_g * x + bb_g;
        const float go = acc[3][r] + wi_o * x + bb_o;
        const size_t idx = (size_t)m * H_ + col;
        const float cn = sigmoidf_(gf) * Cst[idx] + sigmoidf_(gi) * tanhf(gg);
        const float hn = sigmoidf_(go) * tanhf(cn);
        Cst[idx]  = cn;
        Hout[idx] = (__bf16)hn;
        if (t == wlens[m] - 1) lastbf[idx] = (__bf16)hn;
    }
}

// ---------------------------------------------------------------------------
// Plain WMMA GEMM:  D[M,N] = A[M,K_] * B[N,K_]^T  (+ bias) -> bf16
// Used for the embedding projection; same pipelined fragment scheme.
// ---------------------------------------------------------------------------
__global__ __launch_bounds__(256) void wmma_gemm_bias_bf16(
    const __bf16* __restrict__ A, const __bf16* __restrict__ B,
    const float* __restrict__ bias, __bf16* __restrict__ D, int N)
{
    const int lane = threadIdx.x & 31;
    const int wave = threadIdx.x >> 5;
    const int wm   = wave >> 2;
    const int wn   = wave & 3;
    const int m0   = blockIdx.y * 32 + wm * 16;
    const int n0   = blockIdx.x * 256 + wn * 64;
    const int row  = lane & 15;
    const int kh   = lane >> 4;

    const __bf16* Arow = A + (size_t)(m0 + row) * K_;
    const __bf16* Brow[4];
#pragma unroll
    for (int j = 0; j < 4; ++j)
        Brow[j] = B + (size_t)(n0 + j * 16 + row) * K_ + kh * 16;

    v8f acc[4] = {};

    v16bf a_cur = load_afrag(Arow, kh);
    v16bf b_cur[4];
#pragma unroll
    for (int j = 0; j < 4; ++j) b_cur[j] = *(const v16bf*)(Brow[j]);

#pragma unroll
    for (int kb = 0; kb < K_; kb += 32) {
        v16bf a_nxt = a_cur;
        v16bf b_nxt[4];
        if (kb + 32 < K_) {
            a_nxt = load_afrag(Arow + kb + 32, kh);
#pragma unroll
            for (int j = 0; j < 4; ++j) b_nxt[j] = *(const v16bf*)(Brow[j] + kb + 32);
        } else {
#pragma unroll
            for (int j = 0; j < 4; ++j) b_nxt[j] = b_cur[j];
        }
#pragma unroll
        for (int j = 0; j < 4; ++j)
            acc[j] = __builtin_amdgcn_wmma_f32_16x16x32_bf16(
                false, a_cur, false, b_cur[j], (short)0, acc[j], false, false);
        a_cur = a_nxt;
#pragma unroll
        for (int j = 0; j < 4; ++j) b_cur[j] = b_nxt[j];
    }

#pragma unroll
    for (int j = 0; j < 4; ++j) {
        const int col = n0 + j * 16 + row;
        const float bv = bias[col];
#pragma unroll
        for (int r = 0; r < 8; ++r) {
            const int m = m0 + r + 8 * kh;
            D[(size_t)m * N + col] = (__bf16)(acc[j][r] + bv);
        }
    }
}

// ---------------------------------------------------------------------------
// Software grid barrier: one fresh (pre-zeroed) counter per barrier instance.
// ---------------------------------------------------------------------------
__device__ __forceinline__ void grid_barrier(unsigned* cnt, unsigned nblocks) {
    __syncthreads();
    if (threadIdx.x == 0) {
        __threadfence();
        atomicAdd(cnt, 1u);
        while (__hip_atomic_load(cnt, __ATOMIC_ACQUIRE, __HIP_MEMORY_SCOPE_AGENT) < nblocks)
            __builtin_amdgcn_s_sleep(2);
        __threadfence();
    }
    __syncthreads();
}

// ---------------------------------------------------------------------------
// Outer bi-LSTM, one layer. 64 blocks x 256 threads: dir = blk>>5, slice = blk&31.
// Each block computes 64 of the 2048 gates (8 waves x 8 rows; lane-vectorized
// v8bf dots + wave32 shuffle reduction). bf16 weights stay L2-resident.
// ---------------------------------------------------------------------------
__global__ __launch_bounds__(256) void outer_lstm_layer(
    const __bf16* __restrict__ seqin,   // [T][1024] bf16
    __bf16* __restrict__ seqout,        // [T][1024] bf16 (cols dir*512..)
    const __bf16* __restrict__ Wi,      // [2][2048][1024] bf16
    const __bf16* __restrict__ Wh,      // [2][2048][512]  bf16
    const float* __restrict__ bias,     // [2][2048]
    float* __restrict__ gw,             // [2][2048] gate scratch
    float* __restrict__ cs,             // [2][512]
    __bf16* __restrict__ hsv,           // [2][512]
    float* __restrict__ finals,         // [1024]
    unsigned* __restrict__ bars,        // [2][T][2] counters (pre-zeroed)
    int writeFinals)
{
    const int IN = 1024;
    const int dir  = blockIdx.x >> 5;
    const int s    = blockIdx.x & 31;
    const int lane = threadIdx.x & 31;
    const int wave = threadIdx.x >> 5;

    const __bf16* WiD = Wi + (size_t)dir * G_ * IN;
    const __bf16* WhD = Wh + (size_t)dir * G_ * H_;
    const float*  bD  = bias + dir * G_;
    float*        gD  = gw + dir * G_;
    float*        cD  = cs + dir * H_;
    __bf16*       hD  = hsv + dir * H_;
    unsigned*     barD = bars + (size_t)dir * T_ * 2;

    for (int it = 0; it < T_; ++it) {
        const int tt = dir ? (T_ - 1 - it) : it;

        // Cache x (1024) and h (512) slices in registers: lane holds 8-elem chunks.
        v8bf xv[4], hv[2];
        const __bf16* xrow = seqin + (size_t)tt * IN;
#pragma unroll
        for (int i = 0; i < 4; ++i) xv[i] = *(const v8bf*)(xrow + i * 256 + lane * 8);
#pragma unroll
        for (int i = 0; i < 2; ++i) hv[i] = *(const v8bf*)(hD + i * 256 + lane * 8);

        // 8 gate rows per wave
#pragma unroll
        for (int q = 0; q < 8; ++q) {
            const int rowg = s * 64 + wave * 8 + q;
            const __bf16* wi = WiD + (size_t)rowg * IN;
            const __bf16* wh = WhD + (size_t)rowg * H_;
            float acc = 0.f;
#pragma unroll
            for (int i = 0; i < 4; ++i) acc += dot8(*(const v8bf*)(wi + i * 256 + lane * 8), xv[i]);
#pragma unroll
            for (int i = 0; i < 2; ++i) acc += dot8(*(const v8bf*)(wh + i * 256 + lane * 8), hv[i]);
#pragma unroll
            for (int off = 16; off > 0; off >>= 1) acc += __shfl_xor(acc, off, 32);
            if (lane == 0) gD[rowg] = acc + bD[rowg];
        }

        grid_barrier(barD + it * 2 + 0, 32);

        if (threadIdx.x < 16) {
            const int j = s * 16 + (int)threadIdx.x;   // 32 blocks x 16 = 512 hidden units
            float gi = gD[j], gf = gD[H_ + j], gg = gD[2 * H_ + j], go = gD[3 * H_ + j];
            float cn = sigmoidf_(gf) * cD[j] + sigmoidf_(gi) * tanhf(gg);
            float hn = sigmoidf_(go) * tanhf(cn);
            cD[j] = cn;
            hD[j] = (__bf16)hn;
            seqout[(size_t)tt * IN + dir * H_ + j] = (__bf16)hn;
            if (writeFinals && it == T_ - 1) finals[dir * H_ + j] = hn;
        }

        grid_barrier(barD + it * 2 + 1, 32);
    }
}

// ---------------------------------------------------------------------------
// Final FC: out[c] = fc_W[c,:] . concat(final_fwd, final_bwd) + fc_b[c]
// ---------------------------------------------------------------------------
__global__ __launch_bounds__(256) void fc_kernel(
    const float* __restrict__ finals, const float* __restrict__ fcW,
    const float* __restrict__ fcb, float* __restrict__ out)
{
    __shared__ float red[256];
    const int c = blockIdx.x;
    float acc = 0.f;
    for (int i = threadIdx.x; i < 1024; i += 256) acc += fcW[c * 1024 + i] * finals[i];
    red[threadIdx.x] = acc;
    __syncthreads();
    for (int s = 128; s > 0; s >>= 1) {
        if ((int)threadIdx.x < s) red[threadIdx.x] += red[threadIdx.x + s];
        __syncthreads();
    }
    if (threadIdx.x == 0) out[c] = red[0] + fcb[c];
}

// ---------------------------------------------------------------------------
// Host-side orchestration (graph-capturable: kernels + hipMemsetAsync only)
// ---------------------------------------------------------------------------
extern "C" void kernel_launch(void* const* d_in, const int* in_sizes, int n_in,
                              void* d_out, int out_size, void* d_ws, size_t ws_size,
                              hipStream_t stream) {
    (void)in_sizes; (void)n_in; (void)out_size; (void)ws_size;

    const int*   chars = (const int*)d_in[0];
    const int*   wlens = (const int*)d_in[1];
    const float* wWi   = (const float*)d_in[2];
    const float* wWh   = (const float*)d_in[3];
    const float* wb    = (const float*)d_in[4];
    const float* wlW   = (const float*)d_in[5];
    const float* wlb   = (const float*)d_in[6];
    const float* oWi   = (const float*)d_in[7];
    const float* oWh   = (const float*)d_in[8];
    const float* ob    = (const float*)d_in[9];
    const float* fcW   = (const float*)d_in[10];
    const float* fcb   = (const float*)d_in[11];
    float* out = (float*)d_out;

    uint8_t* ws = (uint8_t*)d_ws;
    size_t off = 0;
    auto carve = [&](size_t bytes) { size_t o = off; off += (bytes + 255) & ~(size_t)255; return o; };

    __bf16*  wh_bf   = (__bf16*)(ws + carve((size_t)G_ * H_ * 2));        // 2 MB
    __bf16*  wlw_bf  = (__bf16*)(ws + carve((size_t)E_ * H_ * 2));        // 1 MB
    __bf16*  owi_bf  = (__bf16*)(ws + carve((size_t)4 * G_ * E_ * 2));    // 16 MB
    __bf16*  owh_bf  = (__bf16*)(ws + carve((size_t)4 * G_ * H_ * 2));    // 8 MB
    __bf16*  h_ping  = (__bf16*)(ws + carve((size_t)W_ * H_ * 2));        // 4 MB
    __bf16*  h_pong  = (__bf16*)(ws + carve((size_t)W_ * H_ * 2));        // 4 MB
    float*   c_buf   = (float*) (ws + carve((size_t)W_ * H_ * 4));        // 8 MB
    __bf16*  last_bf = (__bf16*)(ws + carve((size_t)W_ * H_ * 2));        // 4 MB
    __bf16*  seq0    = (__bf16*)(ws + carve((size_t)T_ * E_ * 2));        // 8 MB
    __bf16*  seq1    = (__bf16*)(ws + carve((size_t)T_ * E_ * 2));        // 8 MB
    float*   gatesws = (float*) (ws + carve((size_t)2 * G_ * 4));
    float*   cstate  = (float*) (ws + carve((size_t)2 * H_ * 4));
    __bf16*  hstate  = (__bf16*)(ws + carve((size_t)2 * H_ * 2));
    float*   finals  = (float*) (ws + carve((size_t)2 * H_ * 4));
    unsigned* bars   = (unsigned*)(ws + carve((size_t)2 * T_ * 2 * 4));   // 64 KB

    // --- prep: bf16 weight copies + zeroed LSTM state ---
    hipMemsetAsync(h_ping, 0, (size_t)W_ * H_ * 2, stream);
    hipMemsetAsync(c_buf, 0, (size_t)W_ * H_ * 4, stream);
    cvt_f32_bf16<<<1024, 256, 0, stream>>>(wWh, wh_bf, G_ * H_);
    cvt_f32_bf16<<<1024, 256, 0, stream>>>(wlW, wlw_bf, E_ * H_);
    cvt_f32_bf16<<<2048, 256, 0, stream>>>(oWi, owi_bf, 4 * G_ * E_);
    cvt_f32_bf16<<<2048, 256, 0, stream>>>(oWh, owh_bf, 4 * G_ * H_);

    // --- inner word LSTM: 16 fused WMMA+gate steps (h ping-pong) ---
    dim3 blk(256);
    dim3 grid_lstm(H_ / 64, W_ / 32);      // (8, 128)
    __bf16* hA = h_ping;
    __bf16* hB = h_pong;
    for (int t = 0; t < L_; ++t) {
        wmma_lstm_step<<<grid_lstm, blk, 0, stream>>>(
            hA, wh_bf, chars, wlens, wWi, wb, c_buf, hB, last_bf, t);
        __bf16* tmp = hA; hA = hB; hB = tmp;
    }

    // --- embedding: emb = last @ wl_W^T + wl_b  -> seq0 (bf16) ---
    dim3 grid_emb(E_ / 256, W_ / 32);      // (4, 128)
    wmma_gemm_bias_bf16<<<grid_emb, blk, 0, stream>>>(
        last_bf, wlw_bf, wlb, seq0, E_);

    // --- outer 2-layer bidirectional LSTM (persistent, grid-barriered) ---
    for (int layer = 0; layer < 2; ++layer) {
        hipMemsetAsync(bars, 0, (size_t)2 * T_ * 2 * 4, stream);
        hipMemsetAsync(cstate, 0, (size_t)2 * H_ * 4, stream);
        hipMemsetAsync(hstate, 0, (size_t)2 * H_ * 2, stream);
        const __bf16* sin  = (layer == 0) ? seq0 : seq1;
        __bf16*       sout = (layer == 0) ? seq1 : seq0;   // layer-1 output only feeds finals
        outer_lstm_layer<<<64, 256, 0, stream>>>(
            sin, sout,
            owi_bf + (size_t)layer * 2 * G_ * E_,
            owh_bf + (size_t)layer * 2 * G_ * H_,
            ob + (size_t)layer * 2 * G_,
            gatesws, cstate, hstate, finals, bars, layer == 1);
    }

    // --- classifier head ---
    fc_kernel<<<3, 256, 0, stream>>>(finals, fcW, fcb, out);
}
